// DistributedBrain_58660663328854
// MI455X (gfx1250) — compile-verified, matrix operand order checked by reading, not run on hardware
//
#include <hip/hip_runtime.h>
#include <hip/hip_bf16.h>
#include <math.h>
#include <stdint.h>

// ---------------- problem constants (from reference) ----------------
#define NN   256     // neurons
#define DD   128     // hidden dim
#define VV   32000   // vocab
#define SS   64      // seq len
#define BB   32      // batch
#define KACT 51      // active neurons per step (max(N//5,4))

typedef unsigned short u16;
typedef __attribute__((ext_vector_type(16))) __bf16 v16bf;
typedef __attribute__((ext_vector_type(8)))  float  v8f;

union FragU { uint4 q[2]; v16bf v; };

// round-to-nearest-even f32 -> bf16 (bit pattern)
__device__ inline u16 f2bf(float f) {
    unsigned u = __float_as_uint(f);
    unsigned r = u + 0x7FFFu + ((u >> 16) & 1u);
    return (u16)(r >> 16);
}
// pack two floats into one dword of bf16 (lo, hi)
__device__ inline unsigned f2bf2(float lo, float hi) {
    return ((unsigned)f2bf(hi) << 16) | (unsigned)f2bf(lo);
}

__device__ inline float gelu_exact(float x) {
    return 0.5f * x * (1.0f + erff(x * 0.70710678118654752f));
}

__device__ inline float sigmoidf(float x) {
    return 1.0f / (1.0f + __expf(-x));
}

// Load one 16x32 bf16 fragment from a row-major matrix (stride in elems).
// Per ISA 16-bit A layout: lane L (half=L>>4, r=L&15) holds row (base+r),
// K = {k0 + half*8 + 0..7} and {k0 + 16 + half*8 + 0..7}  -> two b128 loads.
// The same indexing serves B fragments from row-major [N,K] weights
// (lane -> N column, contiguous K).
__device__ inline v16bf frag_ld(const u16* p, int rbase, int k0, int stride) {
    int lane = threadIdx.x & 31;
    int half = lane >> 4;
    int row  = lane & 15;
    const u16* q = p + (size_t)(rbase + row) * stride + k0 + half * 8;
    FragU f;
    f.q[0] = *(const uint4*)(q);
    f.q[1] = *(const uint4*)(q + 16);
    return f.v;
}

__device__ inline v8f wmma_bf16(v16bf a, v16bf b, v8f c) {
    return __builtin_amdgcn_wmma_f32_16x16x32_bf16(
        false, a, false, b, (short)0, c, false, false);
}

// ---------------- prep kernels ----------------
__global__ void cvt_bf16_kernel(const float* __restrict__ src,
                                u16* __restrict__ dst, long n) {
    long i = (long)blockIdx.x * blockDim.x + threadIdx.x;
    if (i < n) dst[i] = f2bf(src[i]);
}

// x[s][b][d] = embed[ids[b][s]][d] + pos[s][d]  (f32 + bf16 copies)
__global__ void embed_kernel(const int* __restrict__ ids,
                             const float* __restrict__ emb,
                             const float* __restrict__ pos,
                             float* __restrict__ x, u16* __restrict__ xbf) {
    int i = blockIdx.x * 256 + threadIdx.x;
    if (i >= SS * BB * DD) return;
    int d = i % DD, b = (i / DD) % BB, s = i / (DD * BB);
    int tok = ids[b * SS + s];
    float v = emb[(size_t)tok * DD + d] + pos[(size_t)s * DD + d];
    size_t o = ((size_t)s * BB + b) * DD + d;
    x[o] = v;
    xbf[o] = f2bf(v);
}

// scores[s][b][n] = x[s][b] . router_w[n] + router_b[n]
// one block per (s,b); x row staged in LDS; thread n owns one neuron row.
__global__ void __launch_bounds__(256)
scores_kernel(const float* __restrict__ x,
              const float* __restrict__ rw,
              const float* __restrict__ rb,
              float* __restrict__ sc) {
    int sb = blockIdx.x;                 // 0 .. SS*BB-1
    __shared__ float xsh[DD];
    int tid = threadIdx.x;
    if (tid < DD) xsh[tid] = x[(size_t)sb * DD + tid];
    __syncthreads();
    int n = tid;                         // NN == 256 == blockDim
    const float4* w4 = (const float4*)(rw + (size_t)n * DD);
    float acc = rb[n];
#pragma unroll
    for (int q = 0; q < DD / 4; ++q) {
        float4 w = w4[q];
        acc += xsh[q * 4 + 0] * w.x + xsh[q * 4 + 1] * w.y +
               xsh[q * 4 + 2] * w.z + xsh[q * 4 + 3] * w.w;
    }
    sc[(size_t)sb * NN + n] = acc;
}

// batch-0 top-K indices per step -> active_idx[s][k], slot[s][n] (-1 inactive)
__global__ void topk0_kernel(const float* __restrict__ sc,
                             int* __restrict__ aidx, int* __restrict__ slot) {
    int s = blockIdx.x * blockDim.x + threadIdx.x;
    if (s >= SS) return;
    for (int n = 0; n < NN; ++n) slot[s * NN + n] = -1;
    const float* row = sc + (size_t)s * BB * NN;   // batch 0
    unsigned used[NN / 32] = {};
    for (int k = 0; k < KACT; ++k) {
        float best = -INFINITY; int bi = 0;
        for (int n = 0; n < NN; ++n) {
            if (used[n >> 5] & (1u << (n & 31))) continue;
            float v = row[n];
            if (v > best) { best = v; bi = n; }
        }
        used[bi >> 5] |= 1u << (bi & 31);
        aidx[s * KACT + k] = bi;
        slot[s * NN + bi] = k;
    }
}

// per (s,b): softmax over that batch's own sorted top-K values -> wsm[s][b][k]
__global__ void topkw_kernel(const float* __restrict__ sc,
                             float* __restrict__ wsm) {
    int i = blockIdx.x * blockDim.x + threadIdx.x;
    if (i >= SS * BB) return;
    int b = i % BB, s = i / BB;
    const float* row = sc + ((size_t)s * BB + b) * NN;
    unsigned used[NN / 32] = {};
    float vals[KACT];
    for (int k = 0; k < KACT; ++k) {
        float best = -INFINITY; int bi = 0;
        for (int n = 0; n < NN; ++n) {
            if (used[n >> 5] & (1u << (n & 31))) continue;
            float v = row[n];
            if (v > best) { best = v; bi = n; }
        }
        used[bi >> 5] |= 1u << (bi & 31);
        vals[k] = best;
    }
    float mx = vals[0], sum = 0.f;
    for (int k = 0; k < KACT; ++k) { vals[k] = __expf(vals[k] - mx); sum += vals[k]; }
    float inv = 1.f / sum;
    for (int k = 0; k < KACT; ++k)
        wsm[((size_t)s * BB + b) * KACT + k] = vals[k] * inv;
}

// ---------------- first hop: states[s][k] = gelu(x@Wt[n]^T+bt)*sig(x.Gw+gb)*w ----------------
__global__ void __launch_bounds__(256)
hop0_kernel(const float* __restrict__ x, const u16* __restrict__ xbf,
            const u16* __restrict__ Wtbf, const float* __restrict__ bt,
            const float* __restrict__ Gw, const float* __restrict__ gb,
            const int* __restrict__ aidx, const float* __restrict__ wsm,
            float* __restrict__ states) {
    int kslot = blockIdx.x, s = blockIdx.y;
    int n = aidx[s * KACT + kslot];
    __shared__ float msh[BB];
    __shared__ float red[256];
    int tid = threadIdx.x;

    // gate: 8 threads per batch row reduce the 128-dot
    {
        int b = tid >> 3, part = tid & 7;
        const float* xb  = x  + ((size_t)s * BB + b) * DD;
        const float* gwn = Gw + (size_t)n * DD;
        float acc = 0.f;
        for (int d = part * 16; d < part * 16 + 16; ++d) acc += xb[d] * gwn[d];
        red[tid] = acc;
        __syncthreads();
        if (part == 0) {
            float t = 0.f;
            for (int j = 0; j < 8; ++j) t += red[b * 8 + j];
            msh[b] = sigmoidf(t + gb[n]) * wsm[((size_t)s * BB + b) * KACT + kslot];
        }
    }
    __syncthreads();

    // GEMM: 8 waves, wave w -> Mtile w>>2, Ntiles {2*(w&3), 2*(w&3)+1}
    int wave = tid >> 5, lane = tid & 31;
    int mt = wave >> 2, nt0 = (wave & 3) * 2;
    int half = lane >> 4, col = lane & 15;
    const u16* xs = xbf  + (size_t)s * BB * DD;
    const u16* wn = Wtbf + (size_t)n * DD * DD;
    v8f c0 = {}, c1 = {};
#pragma unroll
    for (int kb = 0; kb < 4; ++kb) {
        v16bf a  = frag_ld(xs, mt * 16, kb * 32, DD);
        v16bf b0 = frag_ld(wn, nt0 * 16, kb * 32, DD);
        v16bf b1 = frag_ld(wn, (nt0 + 1) * 16, kb * 32, DD);
        c0 = wmma_bf16(a, b0, c0);
        c1 = wmma_bf16(a, b1, c1);
    }
    float* outp = states + ((size_t)s * KACT + kslot) * BB * DD;
    const float* btn = bt + (size_t)n * DD;
#pragma unroll
    for (int v = 0; v < 8; ++v) {
        int brow = mt * 16 + half * 8 + v;
        int d0 = nt0 * 16 + col, d1 = (nt0 + 1) * 16 + col;
        outp[(size_t)brow * DD + d0] = gelu_exact(c0[v] + btn[d0]) * msh[brow];
        outp[(size_t)brow * DD + d1] = gelu_exact(c1[v] + btn[d1]) * msh[brow];
    }
}

// ---------------- edge propagation hop ----------------
// sig = states_in[src] @ We[e]^T ; act = gelu(sig@Wt[dst]^T+bt)*sigmoid(sig.Gw+gb)
// states_out[dst] += 0.5*act  (atomic; states_out pre-copied from states_in)
__global__ void __launch_bounds__(256)
edge_kernel(const float* __restrict__ sin_, float* __restrict__ sout,
            const u16* __restrict__ Webf, const u16* __restrict__ Wtbf,
            const float* __restrict__ bt, const float* __restrict__ Gw,
            const float* __restrict__ gb, const int* __restrict__ slot,
            const int* __restrict__ esrc, const int* __restrict__ edst) {
    int e = blockIdx.x, s = blockIdx.y;
    int src = esrc[e], dst = edst[e];
    int ks = slot[s * NN + src], kd = slot[s * NN + dst];
    if (ks < 0 || kd < 0) return;   // edge not valid this step (block-uniform)

    __shared__ __align__(16) float inf32[BB * DD];   // async-staged f32 tile
    __shared__ __align__(16) u16   inbf[BB * DD];
    __shared__ __align__(16) float sigf[BB * DD];
    __shared__ __align__(16) u16   sigbf[BB * DD];
    __shared__ float msh[BB];
    __shared__ float red[256];

    int tid = threadIdx.x;
    const u16* we = Webf + (size_t)e * DD * DD;
    const u16* wn = Wtbf + (size_t)dst * DD * DD;
    __builtin_prefetch(we, 0, 0);   // global_prefetch_b8: warm weight lines
    __builtin_prefetch(wn, 0, 0);

    // ---- stage states_in[s][ks] (16KB f32) into LDS with async B128 DMA ----
    {
        const float* inp = sin_ + ((size_t)s * KACT + ks) * BB * DD;
        unsigned long long gbase = (unsigned long long)(uintptr_t)inp;
#pragma unroll
        for (int i = tid * 4; i < BB * DD; i += 256 * 4) {
            unsigned loff = (unsigned)(uintptr_t)&inf32[i];   // low 32b = LDS offset
            unsigned goff = (unsigned)(i * sizeof(float));
            asm volatile("global_load_async_to_lds_b128 %0, %1, %2"
                         :: "v"(loff), "v"(goff), "s"(gbase)
                         : "memory");
        }
        asm volatile("s_wait_asynccnt 0x0" ::: "memory");
    }
    __syncthreads();
    // vectorized f32 -> packed bf16 conversion (LDS -> LDS)
#pragma unroll
    for (int i = tid * 4; i < BB * DD; i += 256 * 4) {
        float4 v = *(const float4*)&inf32[i];
        uint2 pk; pk.x = f2bf2(v.x, v.y); pk.y = f2bf2(v.z, v.w);
        *(uint2*)&inbf[i] = pk;
    }
    __syncthreads();

    int wave = tid >> 5, lane = tid & 31;
    int mt = wave >> 2, nt0 = (wave & 3) * 2;
    int half = lane >> 4, col = lane & 15;

    // GEMM1: SIG = IN @ We[e]^T  -> LDS (f32)
    {
        v8f c0 = {}, c1 = {};
#pragma unroll
        for (int kb = 0; kb < 4; ++kb) {
            v16bf a  = frag_ld(inbf, mt * 16, kb * 32, DD);
            v16bf b0 = frag_ld(we, nt0 * 16, kb * 32, DD);
            v16bf b1 = frag_ld(we, (nt0 + 1) * 16, kb * 32, DD);
            c0 = wmma_bf16(a, b0, c0);
            c1 = wmma_bf16(a, b1, c1);
        }
#pragma unroll
        for (int v = 0; v < 8; ++v) {
            int brow = mt * 16 + half * 8 + v;
            sigf[brow * DD + nt0 * 16 + col]       = c0[v];
            sigf[brow * DD + (nt0 + 1) * 16 + col] = c1[v];
        }
    }
    __syncthreads();
    // vectorized f32 -> packed bf16 conversion of SIG
#pragma unroll
    for (int i = tid * 4; i < BB * DD; i += 256 * 4) {
        float4 v = *(const float4*)&sigf[i];
        uint2 pk; pk.x = f2bf2(v.x, v.y); pk.y = f2bf2(v.z, v.w);
        *(uint2*)&sigbf[i] = pk;
    }
    // gate on f32 SIG
    {
        int b = tid >> 3, part = tid & 7;
        const float* gwd = Gw + (size_t)dst * DD;
        float acc = 0.f;
        for (int d = part * 16; d < part * 16 + 16; ++d) acc += sigf[b * DD + d] * gwd[d];
        red[tid] = acc;
        __syncthreads();
        if (part == 0) {
            float t = 0.f;
            for (int j = 0; j < 8; ++j) t += red[b * 8 + j];
            msh[b] = sigmoidf(t + gb[dst]);
        }
    }
    __syncthreads();

    // GEMM2: HE = SIG @ Wt[dst]^T ; epilogue atomic-add into states_out
    {
        const float* btn = bt + (size_t)dst * DD;
        v8f c0 = {}, c1 = {};
#pragma unroll
        for (int kb = 0; kb < 4; ++kb) {
            v16bf a  = frag_ld(sigbf, mt * 16, kb * 32, DD);
            v16bf b0 = frag_ld(wn, nt0 * 16, kb * 32, DD);
            v16bf b1 = frag_ld(wn, (nt0 + 1) * 16, kb * 32, DD);
            c0 = wmma_bf16(a, b0, c0);
            c1 = wmma_bf16(a, b1, c1);
        }
        float* outp = sout + ((size_t)s * KACT + kd) * BB * DD;
#pragma unroll
        for (int v = 0; v < 8; ++v) {
            int brow = mt * 16 + half * 8 + v;
            float m = msh[brow] * 0.5f;
            int d0 = nt0 * 16 + col, d1 = (nt0 + 1) * 16 + col;
            atomicAdd(&outp[(size_t)brow * DD + d0], gelu_exact(c0[v] + btn[d0]) * m);
            atomicAdd(&outp[(size_t)brow * DD + d1], gelu_exact(c1[v] + btn[d1]) * m);
        }
    }
}

// ---------------- aggregate: agg[b*S+s][d] = mean_k states[s][k][b][d] ----------------
__global__ void agg_kernel(const float* __restrict__ states,
                           u16* __restrict__ aggbf) {
    int i = blockIdx.x * 256 + threadIdx.x;
    if (i >= SS * BB * DD) return;
    int d = i % DD, b = (i / DD) % BB, s = i / (DD * BB);
    float acc = 0.f;
    for (int k = 0; k < KACT; ++k)
        acc += states[(((size_t)s * KACT + k) * BB + b) * DD + d];
    aggbf[((size_t)b * SS + s) * DD + d] = f2bf(acc * (1.0f / KACT));
}

// ---------------- output projection: [2048,128] @ [32000,128]^T + bias ----------------
__global__ void __launch_bounds__(256)
outgemm_kernel(const u16* __restrict__ aggbf, const u16* __restrict__ owbf,
               const float* __restrict__ outb, float* __restrict__ out) {
    int n0 = blockIdx.x * 256;   // 125 blocks cover V=32000
    int m0 = blockIdx.y * 32;    // 64 blocks cover 2048 rows
    int tid = threadIdx.x, wave = tid >> 5, lane = tid & 31;
    int mt = wave >> 2, ntbase = (wave & 3) * 4;
    int half = lane >> 4, col = lane & 15;
    v8f acc[4] = {};
#pragma unroll
    for (int kb = 0; kb < 4; ++kb) {
        v16bf a = frag_ld(aggbf, m0 + mt * 16, kb * 32, DD);
#pragma unroll
        for (int t = 0; t < 4; ++t) {
            v16bf bfr = frag_ld(owbf + (size_t)n0 * DD, (ntbase + t) * 16, kb * 32, DD);
            acc[t] = wmma_bf16(a, bfr, acc[t]);
        }
    }
#pragma unroll
    for (int t = 0; t < 4; ++t) {
        int nc = n0 + (ntbase + t) * 16 + col;
        float bias = outb[nc];
#pragma unroll
        for (int v = 0; v < 8; ++v) {
            int r = m0 + mt * 16 + half * 8 + v;   // r = b*S + s
            out[(size_t)r * VV + nc] = acc[t][v] + bias;
        }
    }
}

// ---------------- host launch ----------------
extern "C" void kernel_launch(void* const* d_in, const int* in_sizes, int n_in,
                              void* d_out, int out_size, void* d_ws, size_t ws_size,
                              hipStream_t stream) {
    const int*   ids   = (const int*)  d_in[0];
    const float* emb   = (const float*)d_in[1];
    const float* pos   = (const float*)d_in[2];
    const float* rw    = (const float*)d_in[3];
    const float* rb    = (const float*)d_in[4];
    const float* Wt    = (const float*)d_in[5];
    const float* bt    = (const float*)d_in[6];
    const float* Gw    = (const float*)d_in[7];
    const float* gb    = (const float*)d_in[8];
    const float* We    = (const float*)d_in[9];
    const int*   esrc  = (const int*)  d_in[10];
    const int*   edst  = (const int*)  d_in[11];
    const float* ow    = (const float*)d_in[12];
    const float* ob    = (const float*)d_in[13];
    float* out = (float*)d_out;

    const int E = in_sizes[10];          // number of edges

    // workspace carve-out (256B aligned)
    char* ws = (char*)d_ws;
    size_t off = 0;
    auto alloc = [&](size_t bytes) -> char* {
        char* p = ws + off;
        off = (off + bytes + 255) & ~(size_t)255;
        return p;
    };
    u16*   Wtbf  = (u16*)  alloc((size_t)NN * DD * DD * sizeof(u16));
    u16*   Webf  = (u16*)  alloc((size_t)E  * DD * DD * sizeof(u16));
    u16*   owbf  = (u16*)  alloc((size_t)VV * DD * sizeof(u16));
    float* x     = (float*)alloc((size_t)SS * BB * DD * sizeof(float));
    u16*   xbf   = (u16*)  alloc((size_t)SS * BB * DD * sizeof(u16));
    float* sc    = (float*)alloc((size_t)SS * BB * NN * sizeof(float));
    int*   aidx  = (int*)  alloc((size_t)SS * KACT * sizeof(int));
    int*   slot  = (int*)  alloc((size_t)SS * NN * sizeof(int));
    float* wsm   = (float*)alloc((size_t)SS * BB * KACT * sizeof(float));
    u16*   aggbf = (u16*)  alloc((size_t)BB * SS * DD * sizeof(u16));
    size_t stBytes = (size_t)SS * KACT * BB * DD * sizeof(float);
    float* stA   = (float*)alloc(stBytes);
    float* stB   = (float*)alloc(stBytes);

    // prep: bf16 conversions
    {
        long nWt = (long)NN * DD * DD;
        cvt_bf16_kernel<<<(int)((nWt + 255) / 256), 256, 0, stream>>>(Wt, Wtbf, nWt);
        long nWe = (long)E * DD * DD;
        cvt_bf16_kernel<<<(int)((nWe + 255) / 256), 256, 0, stream>>>(We, Webf, nWe);
        long nOw = (long)VV * DD;
        cvt_bf16_kernel<<<(int)((nOw + 255) / 256), 256, 0, stream>>>(ow, owbf, nOw);
    }
    // embed + pos
    embed_kernel<<<(SS * BB * DD + 255) / 256, 256, 0, stream>>>(ids, emb, pos, x, xbf);
    // router scores (one block per (s,b)), top-k, softmax weights
    scores_kernel<<<SS * BB, 256, 0, stream>>>(x, rw, rb, sc);
    topk0_kernel<<<1, SS, 0, stream>>>(sc, aidx, slot);
    topkw_kernel<<<(SS * BB + 255) / 256, 256, 0, stream>>>(sc, wsm);
    // first hop (all S*K active-neuron GEMMs in parallel)
    hop0_kernel<<<dim3(KACT, SS), 256, 0, stream>>>(x, xbf, Wtbf, bt, Gw, gb,
                                                    aidx, wsm, stA);
    // hop 1: B = A ; edges read A, accumulate into B
    hipMemcpyAsync(stB, stA, stBytes, hipMemcpyDeviceToDevice, stream);
    edge_kernel<<<dim3(E, SS), 256, 0, stream>>>(stA, stB, Webf, Wtbf, bt, Gw, gb,
                                                 slot, esrc, edst);
    // hop 2: A = B ; edges read B, accumulate into A
    hipMemcpyAsync(stA, stB, stBytes, hipMemcpyDeviceToDevice, stream);
    edge_kernel<<<dim3(E, SS), 256, 0, stream>>>(stB, stA, Webf, Wtbf, bt, Gw, gb,
                                                 slot, esrc, edst);
    // aggregate active states -> agg (bf16, row = b*S+s)
    agg_kernel<<<(SS * BB * DD + 255) / 256, 256, 0, stream>>>(stA, aggbf);
    // output projection: 2048 x 32000 x 128
    outgemm_kernel<<<dim3(VV / 256, (BB * SS) / 32), 256, 0, stream>>>(aggbf, owbf, ob, out);

    (void)n_in; (void)out_size; (void)ws_size;
}
